// GraphTransformerBlock_833223656439
// MI455X (gfx1250) — compile-verified, matrix operand order checked by reading, not run on hardware
//
#include <hip/hip_runtime.h>
#include <hip/hip_bf16.h>

// ---- problem constants (from reference setup) ----
#define NB 2
#define NN 8192
#define KN 32
#define CC 128
#define OO 128
#define HS 136              // f16 LDS row stride (halfs) -> conflict-free B128 frag loads
#define FS 132              // f32 LDS row stride (floats)
#define SLOT (128 * HS)     // halfs per LDS slot (34,816 B)

typedef __attribute__((ext_vector_type(16))) _Float16 v16h;
typedef __attribute__((ext_vector_type(8)))  float    v8f;
typedef __attribute__((ext_vector_type(4)))  unsigned int u32x4;
typedef __attribute__((ext_vector_type(8)))  int          i32x8;
typedef __attribute__((ext_vector_type(4)))  int          i32x4;

union Frag16 { v16h h; uint4 u[2]; };
union Acc8   { v8f  v; float f[8]; };
union H8     { _Float16 h[8]; uint4 u; };

#if defined(__has_builtin)
#if __has_builtin(__builtin_amdgcn_tensor_load_to_lds) && __has_builtin(__builtin_amdgcn_s_wait_tensorcnt)
#define USE_TDM 1
#endif
#endif
#ifndef USE_TDM
#define USE_TDM 0
#endif

// ---------------------------------------------------------------------------
// prep: convert the 5 weight matrices to f16 in ws; fold BN into scale/shift
// ---------------------------------------------------------------------------
__global__ void prep_kernel(const float* __restrict__ ipw, const float* __restrict__ opw,
                            const float* __restrict__ cw,  const float* __restrict__ bng,
                            const float* __restrict__ bnb, const float* __restrict__ bnm,
                            const float* __restrict__ bnv,
                            _Float16* __restrict__ w16, float* __restrict__ bnbuf) {
  int t = blockIdx.x * blockDim.x + threadIdx.x;
  if (t < 3 * 16384)      w16[t] = (_Float16)ipw[t];                 // Wq | Wk | Wv
  else if (t < 4 * 16384) w16[t] = (_Float16)opw[t - 3 * 16384];     // Wo
  else if (t < 5 * 16384) w16[t] = (_Float16)cw [t - 4 * 16384];     // conv_w
  if (t < OO) {
    float inv = bng[t] * rsqrtf(bnv[t] + 1e-5f);
    bnbuf[t]      = inv;                       // scale
    bnbuf[OO + t] = bnb[t] - bnm[t] * inv;     // shift
  }
}

// ---------------------------------------------------------------------------
// TDM: DMA one 128x128 f16 weight matrix (compact in ws) into LDS with a
// hardware pad of 4 DWORDs (8 halfs) every 64 DWORDs (128 halfs) -> stride HS.
// Issued by wave 0 only (TDM ignores EXEC; one descriptor per matrix).
// ---------------------------------------------------------------------------
__device__ __forceinline__ void tdm_load_w(const _Float16* src, _Float16* WlLds) {
#if USE_TDM
  unsigned long long ga = (unsigned long long)(size_t)src;
  unsigned lds_off = (unsigned)(size_t)WlLds;   // flat addr low 32 bits == LDS byte offset
  u32x4 g0;
  g0[0] = 1u;                                              // count=1, user descriptor
  g0[1] = lds_off;                                         // lds_addr
  g0[2] = (unsigned)(ga & 0xffffffffu);                    // global_addr[31:0]
  g0[3] = (unsigned)((ga >> 32) & 0x01ffffffu) | (2u << 30); // global_addr[56:32] | type=2
  i32x8 g1;
  g1[0] = (1 << 16)            // data_size = 2 bytes
        | (1 << 20)            // pad_enable
        | (5 << 22)            // pad_interval: 2^(5+1) = 64 DWORDs between pads
        | (3 << 25);           // pad_amount:  4 DWORDs (= 8 halfs) of pad
  g1[1] = (128 << 16);         // tensor_dim0[15:0] = 128      (bits 79:48)
  g1[2] = (128 << 16);         // tensor_dim0 hi=0 | tensor_dim1[15:0]=128 (111:80)
  g1[3] = (128 << 16);         // tensor_dim1 hi=0 | tile_dim0=128 (127:112)
  g1[4] = 128;                 // tile_dim1=128 | tile_dim2=0  (159:128)
  g1[5] = 128;                 // tensor_dim0_stride[31:0] = 128 (207:160)
  g1[6] = 0;                   // stride hi | tensor_dim1_stride lo (unused, 2D tile)
  g1[7] = 0;
  i32x4 z4 = {0, 0, 0, 0};           // groups 2/3: tile_dim3/4 = 0 (unused)
  i32x8 z8 = {0, 0, 0, 0, 0, 0, 0, 0};
  // amdgpu-toolchain (clang-23) 6-arg form
  __builtin_amdgcn_tensor_load_to_lds(g0, g1, z4, z4, z8, 0);
#else
  (void)src; (void)WlLds;
#endif
}

__device__ __forceinline__ void tdm_wait() {
#if USE_TDM
  __builtin_amdgcn_s_wait_tensorcnt(0);
#endif
}

// fallback cooperative staging (only used if TDM builtins are unavailable)
__device__ __forceinline__ void stage_w(_Float16* Wl, const _Float16* src, int tid) {
#if !USE_TDM
  #pragma unroll 4
  for (int i = tid; i < 128 * 16; i += 256) {
    int r = i >> 4, q = i & 15;
    ((uint4*)(Wl + r * HS))[q] = ((const uint4*)src)[r * 16 + q];
  }
#else
  (void)Wl; (void)src; (void)tid;
#endif
}

// ---------------------------------------------------------------------------
// 128x128x128 GEMM: D = A(128x128,f16,LDS) * W^T(col-major-by-out,f16,LDS)
// 8 waves, each wave owns one 16-row group x 128 cols (8 col chunks, 4 k steps)
// MODE 0: D += bias[col], store f16 to dst16
// MODE 1: e = D*bninv[col]+bnsh[col]; leaky-relu; store f32 to dstf
// ---------------------------------------------------------------------------
template <int MODE>
__device__ __forceinline__ void gemm128x128(const _Float16* Al, const _Float16* Wl,
                                            const float* bias, _Float16* dst16,
                                            const float* bninv, const float* bnsh,
                                            float* dstf, int tid) {
  const int lane = tid & 31;
  const int wv   = tid >> 5;
  const int row0 = wv << 4;
  const int l15  = lane & 15;
  const int hi   = (lane & 16) ? 8 : 0;   // ISA 16-bit A-layout: upper half-wave shifts K by 8

  Frag16 af[4];
  #pragma unroll
  for (int kc = 0; kc < 4; ++kc) {
    const _Float16* p = Al + (row0 + l15) * HS + kc * 32 + hi;
    af[kc].u[0] = *(const uint4*)(p);
    af[kc].u[1] = *(const uint4*)(p + 16);
  }
  #pragma unroll
  for (int cc = 0; cc < 8; ++cc) {
    const int col = cc * 16 + l15;
    Acc8 acc;
    const float binit = (MODE == 0) ? bias[col] : 0.0f;
    #pragma unroll
    for (int r = 0; r < 8; ++r) acc.f[r] = binit;
    #pragma unroll
    for (int kc = 0; kc < 4; ++kc) {
      Frag16 bf;
      const _Float16* p = Wl + col * HS + kc * 32 + hi;
      bf.u[0] = *(const uint4*)(p);
      bf.u[1] = *(const uint4*)(p + 16);
      acc.v = __builtin_amdgcn_wmma_f32_16x16x32_f16(false, af[kc].h, false, bf.h,
                                                     (short)0, acc.v, false, false);
    }
    if (MODE == 0) {
      #pragma unroll
      for (int r = 0; r < 8; ++r) {
        int m = row0 + ((lane & 16) ? (r + 8) : r);
        dst16[m * HS + col] = (_Float16)acc.f[r];
      }
    } else {
      float inv = bninv[col], sh = bnsh[col];
      #pragma unroll
      for (int r = 0; r < 8; ++r) {
        int m = row0 + ((lane & 16) ? (r + 8) : r);
        float e = acc.f[r] * inv + sh;
        e = (e >= 0.0f) ? e : 0.2f * e;
        dstf[m * FS + col] = e;
      }
    }
  }
}

// ---------------------------------------------------------------------------
// fully fused block: 2 n-values x both b = 128 rows. gather -> QKV -> attn(2)
// -> out-proj -> conv+BN+LReLU -> softmax over k -> feature reduce -> out
// Weight matrices are TDM double-buffered (W0=S2, W1=S6) so each DMA overlaps
// the previous GEMM's WMMA work.
// ---------------------------------------------------------------------------
__global__ __launch_bounds__(256, 1)
void fused_kernel(const int* __restrict__ gidx, const float* __restrict__ x,
                  const float* __restrict__ feature, const _Float16* __restrict__ w16,
                  const float* __restrict__ ipb, const float* __restrict__ opb,
                  const float* __restrict__ bnbuf, float* __restrict__ out) {
  __shared__ __align__(16) _Float16 sm[7 * SLOT];   // 243,712 B of the 320 KB WGP LDS
  _Float16* S0 = sm;                 // A1 (nf)        -> o        -> e (f32, low half)
  _Float16* S1 = sm + 1 * SLOT;      // A2 (centre-nf)            -> e (f32, high half)
  _Float16* W0 = sm + 2 * SLOT;      // weight buffer 0 (Wq, Wv, conv_w)
  _Float16* S3 = sm + 3 * SLOT;      // q              -> nf_out
  _Float16* S4 = sm + 4 * SLOT;      // k
  _Float16* S5 = sm + 5 * SLOT;      // v
  _Float16* W1 = sm + 6 * SLOT;      // weight buffer 1 (Wk, Wo)
  float* elds = (float*)S0;          // 128 x 132 f32 spans S0..S1

  const int tid   = threadIdx.x;
  const bool w0   = (tid >> 5) == 0;   // wave 0 drives the TDM
  const int n0    = blockIdx.x * 2;

  if (w0) tdm_load_w(w16 + 0 * 16384, W0);          // Wq -> W0 (overlaps gather)
  stage_w(W0, w16 + 0 * 16384, tid);                // (fallback only)

  // prefetch this block's feature cachelines (consumed in the last phase)
  for (int p = tid; p < 512; p += 256) {
    int b = p >> 8, rem = p & 255, o = rem >> 1, g = rem & 1;
    __builtin_prefetch(feature + (((size_t)(b * OO + o) * NN) + n0 + g) * KN, 0, 0);
  }

  // ---- gather: A1 = nf (f16), A2 = centre - nf (f16)
  {
    int row = tid >> 1, c0 = (tid & 1) * 64;
    int b = row >> 6, tok = row & 63, g = tok >> 5, k = tok & 31;
    int n = n0 + g;
    int gi = gidx[((size_t)b * NN + n) * KN + k];
    const float* xs = x + ((size_t)b * NN + gi) * CC + c0;
    const float* xc = x + ((size_t)b * NN + n ) * CC + c0;
    #pragma unroll
    for (int c = 0; c < 64; c += 8) {
      float4 a0 = *(const float4*)(xs + c);
      float4 a1 = *(const float4*)(xs + c + 4);
      float4 b0 = *(const float4*)(xc + c);
      float4 b1 = *(const float4*)(xc + c + 4);
      float av[8] = {a0.x, a0.y, a0.z, a0.w, a1.x, a1.y, a1.z, a1.w};
      float cv[8] = {b0.x, b0.y, b0.z, b0.w, b1.x, b1.y, b1.z, b1.w};
      H8 hn, hd;
      #pragma unroll
      for (int j = 0; j < 8; ++j) {
        hn.h[j] = (_Float16)av[j];
        hd.h[j] = (_Float16)(cv[j] - av[j]);
      }
      *(uint4*)(S0 + row * HS + c0 + c) = hn.u;
      *(uint4*)(S1 + row * HS + c0 + c) = hd.u;
    }
  }
  if (w0) tdm_wait();
  __syncthreads();                                   // Wq ready

  if (w0) tdm_load_w(w16 + 1 * 16384, W1);           // Wk -> W1 (overlaps GEMM q)
  gemm128x128<0>(S0, W0, ipb + 0,   S3, nullptr, nullptr, nullptr, tid);   // q
  if (w0) tdm_wait();
  stage_w(W1, w16 + 1 * 16384, tid);
  __syncthreads();                                   // q stored; Wk ready; W0 free

  if (w0) tdm_load_w(w16 + 2 * 16384, W0);           // Wv -> W0 (overlaps GEMM k)
  gemm128x128<0>(S1, W1, ipb + 128, S4, nullptr, nullptr, nullptr, tid);   // k
  if (w0) tdm_wait();
  stage_w(W0, w16 + 2 * 16384, tid);
  __syncthreads();                                   // k stored; Wv ready; W1 free

  if (w0) tdm_load_w(w16 + 3 * 16384, W1);           // Wo -> W1 (overlaps GEMM v)
  gemm128x128<0>(S0, W0, ipb + 256, S5, nullptr, nullptr, nullptr, tid);   // v
  if (w0) tdm_wait();
  stage_w(W1, w16 + 3 * 16384, tid);
  __syncthreads();                                   // v stored; Wo ready; W0 free

  if (w0) tdm_load_w(w16 + 4 * 16384, W0);           // conv_w -> W0 (overlaps attention)
  // ---- attention: seq len 2 (the outer batch dim), per (token, head)
  if (tid < 128) {
    int tok = tid >> 1, h = tid & 1;
    const _Float16* q0 = S3 + (tok)      * HS + h * 64;
    const _Float16* q1 = S3 + (64 + tok) * HS + h * 64;
    const _Float16* k0 = S4 + (tok)      * HS + h * 64;
    const _Float16* k1 = S4 + (64 + tok) * HS + h * 64;
    const _Float16* v0 = S5 + (tok)      * HS + h * 64;
    const _Float16* v1 = S5 + (64 + tok) * HS + h * 64;
    float s00 = 0, s01 = 0, s10 = 0, s11 = 0;
    #pragma unroll 8
    for (int d = 0; d < 64; ++d) {
      float a0 = (float)q0[d], a1 = (float)q1[d];
      float c0 = (float)k0[d], c1 = (float)k1[d];
      s00 += a0 * c0; s01 += a0 * c1; s10 += a1 * c0; s11 += a1 * c1;
    }
    s00 *= 0.125f; s01 *= 0.125f; s10 *= 0.125f; s11 *= 0.125f;   // 1/sqrt(64)
    float m0 = fmaxf(s00, s01), m1 = fmaxf(s10, s11);
    float e00 = __expf(s00 - m0), e01 = __expf(s01 - m0);
    float e10 = __expf(s10 - m1), e11 = __expf(s11 - m1);
    float r0 = 1.0f / (e00 + e01), r1 = 1.0f / (e10 + e11);
    float w00 = e00 * r0, w01 = e01 * r0, w10 = e10 * r1, w11 = e11 * r1;
    _Float16* o0 = S0 + (tok)      * HS + h * 64;   // A1 slot is dead: reuse for o
    _Float16* o1 = S0 + (64 + tok) * HS + h * 64;
    #pragma unroll 8
    for (int d = 0; d < 64; ++d) {
      float vv0 = (float)v0[d], vv1 = (float)v1[d];
      o0[d] = (_Float16)(w00 * vv0 + w01 * vv1);
      o1[d] = (_Float16)(w10 * vv0 + w11 * vv1);
    }
  }
  if (w0) tdm_wait();
  stage_w(W0, w16 + 4 * 16384, tid);
  __syncthreads();                                   // o in S0; Wo in W1; conv_w in W0

  gemm128x128<0>(S0, W1, opb, S3, nullptr, nullptr, nullptr, tid);         // nf_out
  __syncthreads();

  gemm128x128<1>(S3, W0, nullptr, nullptr, bnbuf, bnbuf + OO, elds, tid);  // e = LReLU(BN(conv))
  __syncthreads();

  // ---- softmax over k (32 neighbors) per (b, n, o), reduce against feature
  for (int task = tid; task < 512; task += 256) {
    int o  = task & 127;
    int bg = task >> 7;
    int b  = bg >> 1, g = bg & 1;
    const float* ec = elds + (b * 64 + g * 32) * FS + o;
    float mx = -3.4e38f;
    #pragma unroll
    for (int k = 0; k < KN; ++k) mx = fmaxf(mx, ec[k * FS]);
    float ex[KN];
    float s = 0.0f;
    #pragma unroll
    for (int k = 0; k < KN; ++k) { float e = __expf(ec[k * FS] - mx); ex[k] = e; s += e; }
    float invs = 1.0f / s;
    const float* fp = feature + (((size_t)(b * OO + o) * NN) + n0 + g) * KN;
    float accum = 0.0f;
    #pragma unroll
    for (int k = 0; k < KN; ++k) accum += ex[k] * fp[k];
    out[((size_t)b * OO + o) * NN + (n0 + g)] = accum * invs;
  }
}

// ---------------------------------------------------------------------------
extern "C" void kernel_launch(void* const* d_in, const int* in_sizes, int n_in,
                              void* d_out, int out_size, void* d_ws, size_t ws_size,
                              hipStream_t stream) {
  const int*   gidx = (const int*)  d_in[0];   // Graph_index (B,N,K) int32
  const float* x    = (const float*)d_in[1];   // (B,N,C) f32
  const float* feat = (const float*)d_in[2];   // (B,O,N,K) f32
  const float* ipw  = (const float*)d_in[3];   // (3C,C)
  const float* ipb  = (const float*)d_in[4];   // (3C,)
  const float* opw  = (const float*)d_in[5];   // (C,C)
  const float* opb  = (const float*)d_in[6];   // (C,)
  const float* cw   = (const float*)d_in[7];   // (O,C)
  const float* bng  = (const float*)d_in[8];
  const float* bnb  = (const float*)d_in[9];
  const float* bnm  = (const float*)d_in[10];
  const float* bnv  = (const float*)d_in[11];
  float* out = (float*)d_out;                  // (B,O,N) f32

  _Float16* w16  = (_Float16*)d_ws;                                   // 5 x 128x128 f16
  float*    bnbf = (float*)((char*)d_ws + (size_t)5 * 16384 * sizeof(_Float16));

  prep_kernel<<<(5 * 16384 + 255) / 256, 256, 0, stream>>>(ipw, opw, cw, bng, bnb, bnm, bnv,
                                                           w16, bnbf);
  fused_kernel<<<NN / 2, 256, 0, stream>>>(gidx, x, feat, w16, ipb, opb, bnbf, out);
}
